// network_3985729650773
// MI455X (gfx1250) — matrix operand order repaired and checked by reading, not compiled, not run on hardware
//
#include <hip/hip_runtime.h>

#define DHALF  128
#define DSTATE 8
#define DTRANK 16
#define DMODEL 256
#define BSZ    16
#define LSEQ   4096
#define NROWS  (BSZ * LSEQ)   // 65536
#define CHUNK  64
#define NCHUNK (LSEQ / CHUNK) // 64

typedef __attribute__((ext_vector_type(16))) __bf16          v16bf;
typedef __attribute__((ext_vector_type(16))) unsigned short  v16u;
typedef __attribute__((ext_vector_type(8)))  unsigned short  v8u;
typedef __attribute__((ext_vector_type(8)))  float           v8f;
typedef __attribute__((ext_vector_type(4)))  float           v4f;

__device__ __forceinline__ unsigned short f2bf_u(float f) {
  union { float f; unsigned u; } c; c.f = f;
  unsigned u = c.u;
  u += 0x7FFFu + ((u >> 16) & 1u);          // round-to-nearest-even bf16
  return (unsigned short)(u >> 16);
}

__device__ __forceinline__ unsigned long long pack4bf(v4f p) {
  return (unsigned long long)f2bf_u(p[0])
       | ((unsigned long long)f2bf_u(p[1]) << 16)
       | ((unsigned long long)f2bf_u(p[2]) << 32)
       | ((unsigned long long)f2bf_u(p[3]) << 48);
}

// ---------------------------------------------------------------------------
// WMMA GEMM:  C[M,ND] = A[M,KD] * W[ND,KD]^T  (dot over K)
//   - W and the 128-row A block staged into LDS as bf16 (one pass over A)
//   - A fragments hoisted into registers (KD/32 x v16u)
//   - two N-tiles in flight (independent WMMA chains hide RAW latency)
//   - LDC compile-time -> immediate-offset stores
// ---------------------------------------------------------------------------
template <int KD, int ND, int KSPLIT, int LDC>
__global__ __launch_bounds__(256) void gemm_wmma(
    const float* __restrict__ A0, int lda0,
    const float* __restrict__ A1, int lda1,
    const float* __restrict__ W,
    float* __restrict__ C) {
  extern __shared__ unsigned short lds[];
  unsigned short* wl = lds;            // ND*KD bf16
  unsigned short* al = lds + ND * KD;  // 128*KD bf16
  const int tid = threadIdx.x;

  // Stage W (contiguous) -> LDS bf16
  constexpr int WCH = (ND * KD) / 4;
  for (int i = tid; i < WCH; i += 256) {
    const int e = i * 4;
    v4f p = *(const v4f*)(W + e);
    *(unsigned long long*)&wl[e] = pack4bf(p);
  }
  // Stage A row-block (128 x KD, split at KSPLIT) -> LDS bf16
  constexpr int ACH = (128 * KD) / 4;
  for (int i = tid; i < ACH; i += 256) {
    const int e = i * 4;
    const int r = e / KD;
    const int k = e & (KD - 1);
    const size_t m = (size_t)blockIdx.x * 128 + r;
    const float* src = (k < KSPLIT) ? (A0 + m * lda0 + k)
                                    : (A1 + m * lda1 + (k - KSPLIT));
    v4f p = *(const v4f*)src;
    *(unsigned long long*)&al[e] = pack4bf(p);
  }
  __syncthreads();

  const int wave  = tid >> 5;
  const int lane  = tid & 31;
  const int lmod  = lane & 15;
  const int hi    = lane >> 4;               // 0: lanes 0-15, 1: lanes 16-31
  const int mbase = blockIdx.x * 128 + wave * 16;
  const unsigned short* arow = al + (wave * 16 + lmod) * KD;

  union V16 { v16u v; v8u h[2]; };
  constexpr int KSTEPS = KD / 32;

  // Hoist all A fragments into registers.
  // A 16x32 bf16 frag: elems 0..15 = K(hi*16..+15)
  V16 afr[KSTEPS];
#pragma unroll
  for (int ks = 0; ks < KSTEPS; ++ks) {
    afr[ks].h[0] = *(const v8u*)(arow + ks * 32 + hi * 16);
    afr[ks].h[1] = *(const v8u*)(arow + ks * 32 + hi * 16 + 8);
  }

#pragma unroll
  for (int nt = 0; nt < ND / 16; nt += 2) {
    v8f acc0 = {}, acc1 = {};
    const unsigned short* w0 = wl + ((nt + 0) * 16 + lmod) * KD;
    const unsigned short* w1 = wl + ((nt + 1) * 16 + lmod) * KD;
#pragma unroll
    for (int ks = 0; ks < KSTEPS; ++ks) {
      // B 32x16 bf16 frag: 16 contiguous K at hi*16
      v16u bv0 = *(const v16u*)(w0 + ks * 32 + hi * 16);
      v16u bv1 = *(const v16u*)(w1 + ks * 32 + hi * 16);
      acc0 = __builtin_amdgcn_wmma_f32_16x16x32_bf16(
          false, __builtin_bit_cast(v16bf, afr[ks].v),
          false, __builtin_bit_cast(v16bf, bv0),
          (short)0, acc0, false, false);
      acc1 = __builtin_amdgcn_wmma_f32_16x16x32_bf16(
          false, __builtin_bit_cast(v16bf, afr[ks].v),
          false, __builtin_bit_cast(v16bf, bv1),
          (short)0, acc1, false, false);
    }
    // C layout: VGPR r, lanes 0-15 -> M=r, lanes 16-31 -> M=8+r; N = lane%16
    float* cb0 = C + (size_t)(mbase + hi * 8) * LDC + nt * 16 + lmod;
#pragma unroll
    for (int r = 0; r < 8; ++r) cb0[(size_t)r * LDC] = acc0[r];
#pragma unroll
    for (int r = 0; r < 8; ++r) cb0[(size_t)r * LDC + 16] = acc1[r];
  }
}

// ---------------------------------------------------------------------------
// Depthwise conv (k=3, SAME, correlation) + SiLU for x and z halves of xz.
// ---------------------------------------------------------------------------
__global__ __launch_bounds__(128) void conv_silu_kernel(
    const float* __restrict__ xz, const float* __restrict__ wx,
    const float* __restrict__ wz, float* __restrict__ xact,
    float* __restrict__ zact) {
  const int c = threadIdx.x;
  const size_t bl = blockIdx.x;
  const int l = (int)(bl & (LSEQ - 1));
  const float* row = xz + bl * DMODEL;

  float xm = (l > 0)        ? row[c - DMODEL]         : 0.f;
  float x0 = row[c];
  float xp = (l < LSEQ - 1) ? row[c + DMODEL]         : 0.f;
  float zm = (l > 0)        ? row[c + DHALF - DMODEL] : 0.f;
  float z0 = row[c + DHALF];
  float zp = (l < LSEQ - 1) ? row[c + DHALF + DMODEL] : 0.f;

  float xa = wx[c * 3 + 0] * xm + wx[c * 3 + 1] * x0 + wx[c * 3 + 2] * xp;
  float za = wz[c * 3 + 0] * zm + wz[c * 3 + 1] * z0 + wz[c * 3 + 2] * zp;
  xa = xa / (1.f + __expf(-xa));
  za = za / (1.f + __expf(-za));
  xact[bl * DHALF + c] = xa;
  zact[bl * DHALF + c] = za;
}

// ---------------------------------------------------------------------------
// delta = softplus(dt @ W_dt^T + b_dt); dt = x_dbl[:, 0:16]
// ---------------------------------------------------------------------------
__global__ __launch_bounds__(256) void delta_kernel(
    const float* __restrict__ xdbl, const float* __restrict__ Wdt,
    const float* __restrict__ bdt, float* __restrict__ delta) {
  const size_t idx = (size_t)blockIdx.x * 256 + threadIdx.x;
  const int d = (int)(idx & (DHALF - 1));
  const size_t row = idx >> 7;
  const float* dtp = xdbl + row * 32;
  float s = bdt[d];
#pragma unroll
  for (int r = 0; r < DTRANK; ++r) s += dtp[r] * Wdt[d * DTRANK + r];
  float v = (s > 20.f) ? s : log1pf(__expf(s));
  delta[row * DHALF + d] = v;
}

// ---------------------------------------------------------------------------
// Chunked parallel scan over L.  h_t = exp(delta_t*A) ∘ h_{t-1} + delta_t*u_t*B_t
// Pass 1: per (b,chunk,d) local end-state (h0=0) and product Π dA.
// ---------------------------------------------------------------------------
__global__ __launch_bounds__(128) void scan_pass1(
    const float* __restrict__ delta, const float* __restrict__ xact,
    const float* __restrict__ xdbl, const float* __restrict__ A_log,
    float* __restrict__ hend, float* __restrict__ prodb) {
  const int bc = blockIdx.x;            // b*NCHUNK + c
  const int b = bc >> 6, c = bc & (NCHUNK - 1);
  const int d = threadIdx.x;

  float Ad[DSTATE];
#pragma unroll
  for (int n = 0; n < DSTATE; ++n) Ad[n] = -__expf(A_log[d * DSTATE + n]);

  float h[DSTATE], p[DSTATE];
#pragma unroll
  for (int n = 0; n < DSTATE; ++n) { h[n] = 0.f; p[n] = 1.f; }

  const size_t l0 = (size_t)b * LSEQ + (size_t)c * CHUNK;
  const float* dptr = delta + l0 * DHALF + d;
  const float* uptr = xact  + l0 * DHALF + d;
  const float* bcp  = xdbl  + l0 * 32 + DTRANK;

  for (int t = 0; t < CHUNK; ++t) {
    __builtin_prefetch(dptr + (size_t)(t + 16) * DHALF, 0, 1);
    __builtin_prefetch(uptr + (size_t)(t + 16) * DHALF, 0, 1);
    const float dl = dptr[(size_t)t * DHALF];
    const float u  = uptr[(size_t)t * DHALF];
    const float du = dl * u;
    const float* bv = bcp + (size_t)t * 32;
#pragma unroll
    for (int n = 0; n < DSTATE; ++n) {
      const float dA = __expf(dl * Ad[n]);
      h[n] = dA * h[n] + du * bv[n];
      p[n] *= dA;
    }
  }
  const size_t o = ((size_t)bc * DHALF + d) * DSTATE;
#pragma unroll
  for (int n = 0; n < DSTATE; ++n) { hend[o + n] = h[n]; prodb[o + n] = p[n]; }
}

// Pass 1.5: sequential scan over the NCHUNK chunk-boundary states.
__global__ __launch_bounds__(128) void scan_mid(
    const float* __restrict__ hend, const float* __restrict__ prodb,
    float* __restrict__ hinit) {
  const int b = blockIdx.x;
  const int d = threadIdx.x;
  float carry[DSTATE];
#pragma unroll
  for (int n = 0; n < DSTATE; ++n) carry[n] = 0.f;
  for (int c = 0; c < NCHUNK; ++c) {
    const size_t o = (((size_t)b * NCHUNK + c) * DHALF + d) * DSTATE;
#pragma unroll
    for (int n = 0; n < DSTATE; ++n) {
      hinit[o + n] = carry[n];
      carry[n] = prodb[o + n] * carry[n] + hend[o + n];
    }
  }
}

// Pass 2: replay each chunk with the corrected initial state; emit y + u*Dp.
__global__ __launch_bounds__(128) void scan_pass2(
    const float* __restrict__ delta, const float* __restrict__ xact,
    const float* __restrict__ xdbl, const float* __restrict__ A_log,
    const float* __restrict__ Dp, const float* __restrict__ hinit,
    float* __restrict__ ybuf) {
  const int bc = blockIdx.x;
  const int b = bc >> 6, c = bc & (NCHUNK - 1);
  const int d = threadIdx.x;

  float Ad[DSTATE];
#pragma unroll
  for (int n = 0; n < DSTATE; ++n) Ad[n] = -__expf(A_log[d * DSTATE + n]);
  const float Dd = Dp[d];

  float h[DSTATE];
  const size_t o = ((size_t)bc * DHALF + d) * DSTATE;
#pragma unroll
  for (int n = 0; n < DSTATE; ++n) h[n] = hinit[o + n];

  const size_t l0 = (size_t)b * LSEQ + (size_t)c * CHUNK;
  const float* dptr = delta + l0 * DHALF + d;
  const float* uptr = xact  + l0 * DHALF + d;
  const float* bcp  = xdbl  + l0 * 32 + DTRANK;
  float*       yo   = ybuf  + l0 * DHALF + d;

  for (int t = 0; t < CHUNK; ++t) {
    __builtin_prefetch(dptr + (size_t)(t + 16) * DHALF, 0, 1);
    __builtin_prefetch(uptr + (size_t)(t + 16) * DHALF, 0, 1);
    const float dl = dptr[(size_t)t * DHALF];
    const float u  = uptr[(size_t)t * DHALF];
    const float du = dl * u;
    const float* bv = bcp + (size_t)t * 32;
    float y = 0.f;
#pragma unroll
    for (int n = 0; n < DSTATE; ++n) {
      const float dA = __expf(dl * Ad[n]);
      h[n] = dA * h[n] + du * bv[n];
      y += h[n] * bv[DSTATE + n];
    }
    yo[(size_t)t * DHALF] = y + u * Dd;
  }
}

// ---------------------------------------------------------------------------
extern "C" void kernel_launch(void* const* d_in, const int* in_sizes, int n_in,
                              void* d_out, int out_size, void* d_ws, size_t ws_size,
                              hipStream_t stream) {
  const float* H      = (const float*)d_in[0];
  const float* W_in   = (const float*)d_in[1];
  const float* Wcx    = (const float*)d_in[2];
  const float* Wcz    = (const float*)d_in[3];
  const float* Wxproj = (const float*)d_in[4];
  const float* Wdt    = (const float*)d_in[5];
  const float* bdt    = (const float*)d_in[6];
  const float* A_log  = (const float*)d_in[7];
  const float* Dp     = (const float*)d_in[8];
  const float* W_out  = (const float*)d_in[9];
  float* out = (float*)d_out;

  char* ws = (char*)d_ws;
  const size_t MB = 1ull << 20;
  float* xz    = (float*)(ws);                 // 64 MB, dead after conv
  float* xact  = (float*)(ws + 64 * MB);       // 32 MB
  float* zact  = (float*)(ws + 96 * MB);       // 32 MB
  float* xdbl  = (float*)(ws + 128 * MB);      //  8 MB
  float* delta = (float*)(ws);                 // alias xz[0:32MB)
  float* ybuf  = (float*)(ws + 32 * MB);       // alias xz[32:64MB)
  float* hend  = (float*)(ws + 136 * MB);      //  4 MB
  float* prodb = (float*)(ws + 140 * MB);      //  4 MB
  float* hinit = (float*)(ws + 144 * MB);      //  4 MB

  const int gemm_grid = NROWS / 128;

  // 1) in_proj: xz = H @ W_in^T   (K=256, N=256); LDS = 128K(W)+64K(A)
  gemm_wmma<256, 256, 256, DMODEL>
      <<<gemm_grid, 256, (256 * 256 + 128 * 256) * 2, stream>>>(
      H, DMODEL, H, DMODEL, W_in, xz);

  // 2) depthwise conv + SiLU
  conv_silu_kernel<<<NROWS, 128, 0, stream>>>(xz, Wcx, Wcz, xact, zact);

  // 3) x_proj: x_dbl = x @ W_xproj^T   (K=128, N=32)
  gemm_wmma<128, 32, 128, 32>
      <<<gemm_grid, 256, (32 * 128 + 128 * 128) * 2, stream>>>(
      xact, DHALF, xact, DHALF, Wxproj, xdbl);

  // 4) delta = softplus(dt @ W_dt^T + b_dt)
  delta_kernel<<<(NROWS * DHALF) / 256, 256, 0, stream>>>(xdbl, Wdt, bdt, delta);

  // 5) chunked parallel scan
  scan_pass1<<<BSZ * NCHUNK, DHALF, 0, stream>>>(delta, xact, xdbl, A_log, hend, prodb);
  scan_mid<<<BSZ, DHALF, 0, stream>>>(hend, prodb, hinit);
  scan_pass2<<<BSZ * NCHUNK, DHALF, 0, stream>>>(delta, xact, xdbl, A_log, Dp, hinit, ybuf);

  // 6) out_proj: out = [y | z] @ W_out^T   (K=256 split at 128, N=256)
  gemm_wmma<256, 256, 128, DMODEL>
      <<<gemm_grid, 256, (256 * 256 + 128 * 256) * 2, stream>>>(
      ybuf, DHALF, zact, DHALF, W_out, out);
}
